// ScaledDotProductAttention_87050397155892
// MI455X (gfx1250) — matrix operand order repaired and checked by reading, not compiled, not run on hardware
//
#include <hip/hip_runtime.h>
#include <hip/hip_bf16.h>

// ---- problem constants (match reference) ----
#define DMODEL 512
#define DKH    64
#define DVH    512
#define NHEAD  8
#define BATCH  2
#define NQS    2048
#define NKS    2048
#define HDK    (NHEAD * DKH)   // 512
#define HDV    (NHEAD * DVH)   // 4096
#define MROWS  (BATCH * NQS)   // 4096
#define LN_EPSF 1e-5f

typedef __attribute__((ext_vector_type(16))) __bf16 v16bf;
typedef __attribute__((ext_vector_type(8)))  float  v8f;

union Frag {
  v16bf v;
  __bf16 e[16];
  uint4  q2[2];
};

union Pack8 {
  __bf16 e[8];
  uint4  u;
};

__device__ __forceinline__ v8f wmma_bf16(v16bf a, v16bf b, v8f c) {
  // D = A(16x32 bf16) * B(32x16 bf16) + C(16x16 f32)
  return __builtin_amdgcn_wmma_f32_16x16x32_bf16(
      /*neg_a=*/false, a, /*neg_b=*/false, b,
      /*c_mod=*/(short)0, c, /*reuse_a=*/false, /*reuse_b=*/false);
}

// A operand: row-major (16 x 32) tile at `base`, row stride in elements.
// ISA layout: lanes 0-15 hold M=lane, K {0..7,16..23}; lanes 16-31 K {8..15,24..31}.
__device__ __forceinline__ v16bf ld_a(const __bf16* __restrict__ base, int row_stride, int lane) {
  const int half = lane >> 4;
  const __bf16* p = base + (size_t)(lane & 15) * row_stride + half * 8;
  Frag f;
  f.q2[0] = *(const uint4*)(p);        // K chunk 0..7 (or 8..15)
  f.q2[1] = *(const uint4*)(p + 16);   // K chunk 16..23 (or 24..31)
  return f.v;
}

// B operand: row-major (32 x 16) tile at `base`; lane%16 == N column, K {0..7,16..23} / {8..15,24..31} per half.
__device__ __forceinline__ v16bf ld_b(const __bf16* __restrict__ base, int row_stride, int lane) {
  const int half = lane >> 4;
  const __bf16* p = base + (lane & 15) + (size_t)(half * 8) * row_stride;
  Frag f;
#pragma unroll
  for (int s = 0; s < 8; ++s) {
    f.e[s]     = p[(size_t)s * row_stride];
    f.e[s + 8] = p[(size_t)(16 + s) * row_stride];
  }
  return f.v;
}

// One k-step worth of operands for a 32x64 wave tile.
struct FragSet {
  v16bf a0, a1;
  v16bf b[4];
};

__device__ __forceinline__ void load_set(FragSet& f,
                                         const __bf16* __restrict__ A, int lda,
                                         const __bf16* __restrict__ B, int ldb,
                                         int k0, int lane) {
  f.a0 = ld_a(A + k0, lda, lane);
  f.a1 = ld_a(A + (size_t)16 * lda + k0, lda, lane);
#pragma unroll
  for (int t = 0; t < 4; ++t)
    f.b[t] = ld_b(B + (size_t)k0 * ldb + t * 16, ldb, lane);
}

__device__ __forceinline__ void compute_set(const FragSet& f, v8f (&acc)[2][4]) {
#pragma unroll
  for (int t = 0; t < 4; ++t) {
    acc[0][t] = wmma_bf16(f.a0, f.b[t], acc[0][t]);
    acc[1][t] = wmma_bf16(f.a1, f.b[t], acc[1][t]);
  }
}

// Core 32x64 wave tile with register double-buffering:
// acc[2][4] += A(32 x K) @ B(K x 64).  K/32 must be even (true for all call sites).
__device__ __forceinline__ void mm_32x64(const __bf16* __restrict__ A, int lda,
                                         const __bf16* __restrict__ B, int ldb,
                                         int K, int lane, v8f (&acc)[2][4]) {
  FragSet f0, f1;
  load_set(f0, A, lda, B, ldb, 0, lane);
  for (int k0 = 0; k0 < K; k0 += 64) {
    if (k0 + 128 < K) {  // WGP-scope prefetch two steps ahead -> global_prefetch_b8
      __builtin_prefetch(A + (size_t)(lane & 15) * lda + k0 + 128, 0, 3);
      __builtin_prefetch(B + (size_t)(k0 + 128 + lane) * ldb, 0, 3);
    }
    load_set(f1, A, lda, B, ldb, k0 + 32, lane);   // in flight during compute(f0)
    compute_set(f0, acc);
    if (k0 + 64 < K)
      load_set(f0, A, lda, B, ldb, k0 + 64, lane); // in flight during compute(f1)
    compute_set(f1, acc);
  }
}

// ---------------- f32 -> bf16 conversion ----------------
__global__ void cvt_bf16_kernel(const float* __restrict__ src, __bf16* __restrict__ dst, int n) {
  int i = blockIdx.x * blockDim.x + threadIdx.x;
  if (i < n) dst[i] = (__bf16)src[i];
}

// ---------------- projection GEMM: C(bf16) = A(M,K) @ B(K,N) + bias ----------------
// block = 256 threads = 8 waves; block tile 256(M) x 64(N); wave tile 32 x 64.
// transposeC==0: C[m*N + n]   (row-major, (b, token, feature))
// transposeC==1: Ct[(b*N + n)*NQS + j]  (feature-major per batch, token j contiguous)
__global__ __launch_bounds__(256)
void proj_gemm_kernel(const __bf16* __restrict__ A, const __bf16* __restrict__ B,
                      const float* __restrict__ bias, __bf16* __restrict__ C,
                      int N, int K, int transposeC) {
  const int lane = threadIdx.x & 31;
  const int wave = threadIdx.x >> 5;
  const int m0 = blockIdx.y * 256 + wave * 32;
  const int n0 = blockIdx.x * 64;
  v8f acc[2][4] = {};
  mm_32x64(A + (size_t)m0 * K, K, B + n0, N, K, lane, acc);

  const int half = lane >> 4;
  const int ncol = n0 + (lane & 15);
  if (!transposeC) {
#pragma unroll
    for (int mt = 0; mt < 2; ++mt) {
      const int mrow = m0 + mt * 16 + half * 8;
#pragma unroll
      for (int t = 0; t < 4; ++t) {
        const int n = ncol + t * 16;
        const float bn = bias[n];
#pragma unroll
        for (int r = 0; r < 8; ++r)
          C[(size_t)(mrow + r) * N + n] = (__bf16)(acc[mt][t][r] + bn);
      }
    }
  } else {
    // per lane: 8 consecutive tokens j for a fixed feature n -> one 16B store
    const int b = m0 / NQS;           // block tile (256) never crosses a batch boundary
#pragma unroll
    for (int mt = 0; mt < 2; ++mt) {
      const int j0 = (m0 - b * NQS) + mt * 16 + half * 8;
#pragma unroll
      for (int t = 0; t < 4; ++t) {
        const int n = ncol + t * 16;
        const float bn = bias[n];
        Pack8 pk;
#pragma unroll
        for (int r = 0; r < 8; ++r) pk.e[r] = (__bf16)(acc[mt][t][r] + bn);
        *(uint4*)(C + ((size_t)b * N + n) * NQS + j0) = pk.u;
      }
    }
  }
}

// ---------------- S = (Q K^T / 8) * attw, masked -> f32 ----------------
// kt is feature-major: kt[(b*HDK + d)*NKS + j]
__global__ __launch_bounds__(256)
void qk_score_kernel(const __bf16* __restrict__ qb, const __bf16* __restrict__ kt,
                     const float* __restrict__ attw, const unsigned char* __restrict__ mask,
                     float* __restrict__ S) {
  const int lane = threadIdx.x & 31;
  const int wave = threadIdx.x >> 5;
  const int z = blockIdx.z;                 // b*NHEAD + h
  const int b = z >> 3, h = z & 7;
  const int m0 = blockIdx.y * 256 + wave * 32;   // query rows
  const int n0 = blockIdx.x * 64;                // key cols
  const __bf16* Abase = qb + ((size_t)b * NQS + m0) * HDK + h * DKH;
  const __bf16* Bbase = kt + ((size_t)b * HDK + h * DKH) * NKS + n0;
  v8f acc[2][4] = {};
  mm_32x64(Abase, HDK, Bbase, NKS, DKH, lane, acc);

  const float NEG_INF = -__builtin_inff();
  const int half = lane >> 4;
  const int kcol = n0 + (lane & 15);
#pragma unroll
  for (int mt = 0; mt < 2; ++mt) {
    const int qrow = m0 + mt * 16 + half * 8;
#pragma unroll
    for (int t = 0; t < 4; ++t) {
      const int kk = kcol + t * 16;
#pragma unroll
      for (int r = 0; r < 8; ++r) {
        const int qq = qrow + r;
        const size_t sidx = ((size_t)z * NQS + qq) * NKS + kk;
        const size_t midx = ((size_t)b * NQS + qq) * NKS + kk;
        const float s = acc[mt][t][r] * 0.125f * attw[sidx];
        S[sidx] = mask[midx] ? s : NEG_INF;
      }
    }
  }
}

// ---------------- softmax over the q axis (axis=-2): one thread per (b,h,k) column ----------------
__global__ __launch_bounds__(256)
void softmax_q_kernel(const float* __restrict__ S, __bf16* __restrict__ P) {
  const int col = blockIdx.x * blockDim.x + threadIdx.x;  // over BATCH*NHEAD*NKS
  const int z = col / NKS;
  const int kk = col - z * NKS;
  const float* s = S + (size_t)z * NQS * NKS + kk;
  // online max + sum in one pass
  float m = -__builtin_inff();
  float sum = 0.f;
  for (int q = 0; q < NQS; ++q) {
    const float v = s[(size_t)q * NKS];
    if (v > m) { sum *= __expf(m - v); m = v; }
    sum += __expf(v - m);
  }
  const float inv = 1.0f / sum;
  __bf16* p = P + (size_t)z * NQS * NKS + kk;
  for (int q = 0; q < NQS; ++q)
    p[(size_t)q * NKS] = (__bf16)(__expf(s[(size_t)q * NKS] - m) * inv);
}

// ---------------- attn(b,q,h*DVH+d) = P(b,h,q,:) @ V(b,:,h*DVH+d) ----------------
__global__ __launch_bounds__(256)
void pv_gemm_kernel(const __bf16* __restrict__ P, const __bf16* __restrict__ vb,
                    __bf16* __restrict__ attn) {
  const int lane = threadIdx.x & 31;
  const int wave = threadIdx.x >> 5;
  const int z = blockIdx.z;
  const int b = z >> 3, h = z & 7;
  const int m0 = blockIdx.y * 256 + wave * 32;   // query rows
  const int n0 = blockIdx.x * 64;                // d cols within DVH
  const __bf16* Abase = P + (size_t)z * NQS * NKS + (size_t)m0 * NKS;
  const __bf16* Bbase = vb + (size_t)b * NKS * HDV + h * DVH + n0;
  v8f acc[2][4] = {};
  mm_32x64(Abase, NKS, Bbase, HDV, NKS, lane, acc);

  const int half = lane >> 4;
  const int dcol = h * DVH + n0 + (lane & 15);
#pragma unroll
  for (int mt = 0; mt < 2; ++mt) {
    const int qrow = m0 + mt * 16 + half * 8;
#pragma unroll
    for (int t = 0; t < 4; ++t)
#pragma unroll
      for (int r = 0; r < 8; ++r)
        attn[((size_t)b * NQS + qrow + r) * HDV + dcol + t * 16] = (__bf16)acc[mt][t][r];
  }
}

// ---------------- X = attn @ Wo + bo + queries (residual), f32 ----------------
__global__ __launch_bounds__(256)
void out_gemm_kernel(const __bf16* __restrict__ attn, const __bf16* __restrict__ Wo,
                     const float* __restrict__ bo, const float* __restrict__ queries,
                     float* __restrict__ X) {
  const int lane = threadIdx.x & 31;
  const int wave = threadIdx.x >> 5;
  const int m0 = blockIdx.y * 256 + wave * 32;   // over MROWS
  const int n0 = blockIdx.x * 64;
  v8f acc[2][4] = {};
  mm_32x64(attn + (size_t)m0 * HDV, HDV, Wo + n0, DMODEL, HDV, lane, acc);

  const int half = lane >> 4;
  const int ncol = n0 + (lane & 15);
#pragma unroll
  for (int mt = 0; mt < 2; ++mt) {
    const int mrow = m0 + mt * 16 + half * 8;
#pragma unroll
    for (int t = 0; t < 4; ++t) {
      const int n = ncol + t * 16;
      const float bn = bo[n];
#pragma unroll
      for (int r = 0; r < 8; ++r) {
        const size_t idx = (size_t)(mrow + r) * DMODEL + n;
        X[idx] = acc[mt][t][r] + bn + queries[idx];
      }
    }
  }
}

// ---------------- LayerNorm over last dim (512) ----------------
__global__ __launch_bounds__(256)
void layernorm_kernel(const float* __restrict__ X, const float* __restrict__ gamma,
                      const float* __restrict__ beta, float* __restrict__ out) {
  __shared__ float ssum[256];
  __shared__ float ssq[256];
  const int row = blockIdx.x;
  const int tid = threadIdx.x;
  const float* x = X + (size_t)row * DMODEL;
  const float v0 = x[tid], v1 = x[tid + 256];
  ssum[tid] = v0 + v1;
  ssq[tid]  = v0 * v0 + v1 * v1;
  __syncthreads();
  for (int s = 128; s > 0; s >>= 1) {
    if (tid < s) { ssum[tid] += ssum[tid + s]; ssq[tid] += ssq[tid + s]; }
    __syncthreads();
  }
  const float mu  = ssum[0] * (1.0f / DMODEL);
  const float var = ssq[0] * (1.0f / DMODEL) - mu * mu;
  const float inv = rsqrtf(var + LN_EPSF);
  float* o = out + (size_t)row * DMODEL;
  o[tid]       = gamma[tid]       * (v0 - mu) * inv + beta[tid];
  o[tid + 256] = gamma[tid + 256] * (v1 - mu) * inv + beta[tid + 256];
}

extern "C" void kernel_launch(void* const* d_in, const int* in_sizes, int n_in,
                              void* d_out, int out_size, void* d_ws, size_t ws_size,
                              hipStream_t stream) {
  (void)in_sizes; (void)n_in; (void)out_size; (void)ws_size;
  const float* queries = (const float*)d_in[0];
  const float* keys    = (const float*)d_in[1];
  const float* values  = (const float*)d_in[2];
  const unsigned char* mask = (const unsigned char*)d_in[3];  // numpy bool, 1 byte
  const float* attw = (const float*)d_in[4];
  const float* Wq = (const float*)d_in[5];
  const float* bq = (const float*)d_in[6];
  const float* Wk = (const float*)d_in[7];
  const float* bk = (const float*)d_in[8];
  const float* Wv = (const float*)d_in[9];
  const float* bv = (const float*)d_in[10];
  const float* Wo = (const float*)d_in[11];
  const float* bo = (const float*)d_in[12];
  const float* gamma = (const float*)d_in[13];
  const float* beta  = (const float*)d_in[14];
  float* out = (float*)d_out;

  char* ws = (char*)d_ws;
  size_t off = 0;
  auto take = [&](size_t bytes) -> char* {
    char* p = ws + off;
    off += (bytes + 255) & ~(size_t)255;
    return p;
  };
  __bf16* xq_bf = (__bf16*)take((size_t)MROWS * DMODEL * 2);   // queries bf16
  __bf16* xk_bf = (__bf16*)take((size_t)MROWS * DMODEL * 2);   // keys bf16
  __bf16* xv_bf = (__bf16*)take((size_t)MROWS * DMODEL * 2);   // values bf16
  __bf16* wq_bf = (__bf16*)take((size_t)DMODEL * HDK * 2);
  __bf16* wk_bf = (__bf16*)take((size_t)DMODEL * HDK * 2);
  __bf16* wv_bf = (__bf16*)take((size_t)DMODEL * HDV * 2);
  __bf16* wo_bf = (__bf16*)take((size_t)HDV * DMODEL * 2);
  __bf16* qb    = (__bf16*)take((size_t)MROWS * HDK * 2);      // Q proj (token-major)
  __bf16* kt    = (__bf16*)take((size_t)MROWS * HDK * 2);      // K proj (feature-major)
  __bf16* vb    = (__bf16*)take((size_t)MROWS * HDV * 2);      // V proj (token-major)
  float*  S     = (float*) take((size_t)BATCH * NHEAD * NQS * NKS * 4);
  __bf16* P     = (__bf16*)take((size_t)BATCH * NHEAD * NQS * NKS * 2);
  __bf16* attn  = (__bf16*)take((size_t)MROWS * HDV * 2);
  float*  X     = (float*) take((size_t)MROWS * DMODEL * 4);

  auto cvt = [&](const float* s, __bf16* d, int n) {
    cvt_bf16_kernel<<<dim3((unsigned)((n + 255) / 256)), dim3(256), 0, stream>>>(s, d, n);
  };
  cvt(queries, xq_bf, MROWS * DMODEL);
  cvt(keys,    xk_bf, MROWS * DMODEL);
  cvt(values,  xv_bf, MROWS * DMODEL);
  cvt(Wq, wq_bf, DMODEL * HDK);
  cvt(Wk, wk_bf, DMODEL * HDK);
  cvt(Wv, wv_bf, DMODEL * HDV);
  cvt(Wo, wo_bf, HDV * DMODEL);

  // Q/K/V projections (K written feature-major so QK^T needs no transposed gather)
  proj_gemm_kernel<<<dim3(HDK / 64, MROWS / 256), dim3(256), 0, stream>>>(xq_bf, wq_bf, bq, qb, HDK, DMODEL, 0);
  proj_gemm_kernel<<<dim3(HDK / 64, MROWS / 256), dim3(256), 0, stream>>>(xk_bf, wk_bf, bk, kt, HDK, DMODEL, 1);
  proj_gemm_kernel<<<dim3(HDV / 64, MROWS / 256), dim3(256), 0, stream>>>(xv_bf, wv_bf, bv, vb, HDV, DMODEL, 0);

  // S = (QK^T / sqrt(64)) * attw, masked
  qk_score_kernel<<<dim3(NKS / 64, NQS / 256, BATCH * NHEAD), dim3(256), 0, stream>>>(qb, kt, attw, mask, S);

  // softmax over query axis -> P (bf16)
  softmax_q_kernel<<<dim3((BATCH * NHEAD * NKS) / 256), dim3(256), 0, stream>>>(S, P);

  // attn = P @ V  (dominant GEMM)
  pv_gemm_kernel<<<dim3(DVH / 64, NQS / 256, BATCH * NHEAD), dim3(256), 0, stream>>>(P, vb, attn);

  // output projection + bias + residual
  out_gemm_kernel<<<dim3(DMODEL / 64, MROWS / 256), dim3(256), 0, stream>>>(attn, wo_bf, bo, queries, X);

  // LayerNorm
  layernorm_kernel<<<dim3(MROWS), dim3(256), 0, stream>>>(X, gamma, beta, out);
}
